// MultiHeadAttention_57294863729336
// MI455X (gfx1250) — compile-verified, matrix operand order checked by reading, not run on hardware
//
#include <hip/hip_runtime.h>
#include <hip/hip_bf16.h>

typedef __bf16 bf16_t;
typedef __attribute__((ext_vector_type(16))) __bf16 v16bf;
typedef __attribute__((ext_vector_type(8)))  __bf16 v8bf;
typedef __attribute__((ext_vector_type(8)))  float  v8f;

#define ATTN_SCALE 0.125f   // 64^-0.5

// ---------------------------------------------------------------------------
// Fragment loader for the CDNA5 WMMA 16-bit A/B layout (16x32, K=32).
// Lane L holds row (row_base + (L&15)); K-run offset 0 or 8 depending on the
// lane half; two contiguous 8-element (16-byte) runs at k0+koff and k0+16+koff.
// Works for A (row-major MxK) and for B given as B^T row-major (NxK).
// ---------------------------------------------------------------------------
__device__ inline v16bf load_frag(const bf16_t* __restrict__ base, int ld,
                                  int row_base, int k0) {
  const int lane = threadIdx.x & 31;
  const int r    = row_base + (lane & 15);
  const int koff = (lane & 16) >> 1;            // 0 or 8
  const bf16_t* p = base + (long)r * ld + k0 + koff;
  v8bf lo = *(const v8bf*)(p);
  v8bf hi = *(const v8bf*)(p + 16);
  return __builtin_shufflevector(lo, hi, 0,1,2,3,4,5,6,7,8,9,10,11,12,13,14,15);
}

__device__ inline v8f wmma_bf16(v16bf a, v16bf b, v8f c) {
  return __builtin_amdgcn_wmma_f32_16x16x32_bf16(
      /*neg_a=*/false, a, /*neg_b=*/false, b,
      /*c_mod=*/(short)0, c, /*reuse_a=*/false, /*reuse_b=*/false);
}

// ---------------------------------------------------------------------------
// f32 -> bf16 elementwise convert
// ---------------------------------------------------------------------------
__global__ void mha_f32_to_bf16(const float* __restrict__ in,
                                bf16_t* __restrict__ out, int n) {
  int i = blockIdx.x * blockDim.x + threadIdx.x;
  if (i < n) out[i] = (bf16_t)in[i];
}

// ---------------------------------------------------------------------------
// QKV projection: Y[m,n] = sum_k X[m,k] * W[n,k]   (M=4096, N=768, K=768)
// One wave -> 32x64 output block (2x4 WMMA tiles): 8 WMMAs per k-step against
// 6 fragment loads (A frags reused 4x, B frags reused 2x).
// vtrans==0: store [b][h][seq][d] bf16 ;  vtrans==1: store [b][h][d][seq] bf16
// ---------------------------------------------------------------------------
__global__ __launch_bounds__(128) void mha_qkv_gemm(
    const bf16_t* __restrict__ X, const bf16_t* __restrict__ W,
    bf16_t* __restrict__ out, int vtrans) {
  const int wave = threadIdx.x >> 5;
  const int lane = threadIdx.x & 31;
  const int m0 = blockIdx.x * 32;                 // 128 m-blocks of 32
  const int n0 = (blockIdx.y * 4 + wave) * 64;    // 12 n-blocks of 64

  v8f acc00 = {}, acc01 = {}, acc02 = {}, acc03 = {};
  v8f acc10 = {}, acc11 = {}, acc12 = {}, acc13 = {};

  for (int k = 0; k < 768; k += 32) {
    v16bf a0 = load_frag(X, 768, m0,      k);
    v16bf a1 = load_frag(X, 768, m0 + 16, k);
    v16bf b0 = load_frag(W, 768, n0,      k);
    v16bf b1 = load_frag(W, 768, n0 + 16, k);
    v16bf b2 = load_frag(W, 768, n0 + 32, k);
    v16bf b3 = load_frag(W, 768, n0 + 48, k);
    acc00 = wmma_bf16(a0, b0, acc00);
    acc01 = wmma_bf16(a0, b1, acc01);
    acc02 = wmma_bf16(a0, b2, acc02);
    acc03 = wmma_bf16(a0, b3, acc03);
    acc10 = wmma_bf16(a1, b0, acc10);
    acc11 = wmma_bf16(a1, b1, acc11);
    acc12 = wmma_bf16(a1, b2, acc12);
    acc13 = wmma_bf16(a1, b3, acc13);
  }

  const int hi = (lane & 16) ? 8 : 0;
  v8f* accs[2][4] = {{&acc00, &acc01, &acc02, &acc03},
                     {&acc10, &acc11, &acc12, &acc13}};
#pragma unroll
  for (int mi = 0; mi < 2; ++mi) {
#pragma unroll
    for (int ni = 0; ni < 4; ++ni) {
      const v8f a = *accs[mi][ni];
      const int n_g = n0 + ni * 16 + (lane & 15);
      const int h = n_g >> 6, d = n_g & 63;
#pragma unroll
      for (int j = 0; j < 8; ++j) {
        const int m_g = m0 + mi * 16 + j + hi;
        const int b_ = m_g >> 11, seq = m_g & 2047;
        long idx;
        if (vtrans) idx = ((long)(b_ * 12 + h) * 64 + d) * 2048 + seq;
        else        idx = ((long)(b_ * 12 + h) * 2048 + seq) * 64 + d;
        out[idx] = (bf16_t)a[j];
      }
    }
  }
}

// ---------------------------------------------------------------------------
// Flash attention: one wave owns 16 query rows of one (b,h); streams keys in
// 32-column chunks with online softmax. Per chunk: 4 WMMA (S) + 4 WMMA (PV).
// Q,K: [bh][seq][64] bf16 ; Vt: [bh][64][seq] bf16 ; AO: [b][seq][768] bf16
// ---------------------------------------------------------------------------
__global__ __launch_bounds__(128) void mha_flash_attn(
    const bf16_t* __restrict__ Q, const bf16_t* __restrict__ K,
    const bf16_t* __restrict__ Vt, const int* __restrict__ mask,
    bf16_t* __restrict__ AO) {
  __shared__ __align__(16) bf16_t plds[4][16 * 32];

  const int wave = threadIdx.x >> 5;
  const int lane = threadIdx.x & 31;
  const int gw   = blockIdx.x * 4 + wave;   // 0..3071
  const int bh   = gw >> 7;                 // 0..23
  const int mtile= gw & 127;
  const int b    = bh / 12, h = bh % 12;
  const int hi   = (lane & 16) ? 8 : 0;
  const int m0   = mtile * 16;

  const bf16_t* q  = Q  + (long)bh * 2048 * 64;
  const bf16_t* kk = K  + (long)bh * 2048 * 64;
  const bf16_t* vt = Vt + (long)bh * 64 * 2048;
  const int* mrow  = mask + b * 2048;
  bf16_t* myp = plds[wave];

  const v16bf q0 = load_frag(q, 64, m0, 0);
  const v16bf q1 = load_frag(q, 64, m0, 32);

  float mrun[8], lrun[8];
  int rm[8];
  v8f o0 = {}, o1 = {}, o2 = {}, o3 = {};
#pragma unroll
  for (int j = 0; j < 8; ++j) {
    mrun[j] = -3.0e30f; lrun[j] = 0.0f;
    rm[j] = mrow[m0 + j + hi];
  }

  for (int n = 0; n < 2048; n += 32) {
    // S = q @ k^T for key columns [n, n+32)
    v16bf b0a = load_frag(kk, 64, n,      0);
    v16bf b0b = load_frag(kk, 64, n,      32);
    v16bf b1a = load_frag(kk, 64, n + 16, 0);
    v16bf b1b = load_frag(kk, 64, n + 16, 32);
    v8f s0 = {}, s1 = {};
    s0 = wmma_bf16(q0, b0a, s0); s0 = wmma_bf16(q1, b0b, s0);
    s1 = wmma_bf16(q0, b1a, s1); s1 = wmma_bf16(q1, b1b, s1);

    const int mc0 = mrow[n +      (lane & 15)];
    const int mc1 = mrow[n + 16 + (lane & 15)];

    float p0[8], p1[8], alpha[8];
#pragma unroll
    for (int j = 0; j < 8; ++j) {
      float a = s0[j], c = s1[j];
      a = (rm[j] && mc0) ? a * ATTN_SCALE : -1000.0f;
      c = (rm[j] && mc1) ? c * ATTN_SCALE : -1000.0f;
      // row max over this 32-wide chunk (butterfly within 16-lane half)
      float cm = fmaxf(a, c);
      for (int off = 1; off < 16; off <<= 1)
        cm = fmaxf(cm, __shfl_xor(cm, off, 32));
      const float nm = fmaxf(mrun[j], cm);
      alpha[j] = __expf(mrun[j] - nm);
      const float e0 = __expf(a - nm);
      const float e1 = __expf(c - nm);
      float rs = e0 + e1;
      for (int off = 1; off < 16; off <<= 1)
        rs += __shfl_xor(rs, off, 32);
      lrun[j] = lrun[j] * alpha[j] + rs;
      mrun[j] = nm;
      p0[j] = e0; p1[j] = e1;
    }

    // rescale accumulators + stage P tile (16x32 bf16) through LDS
#pragma unroll
    for (int j = 0; j < 8; ++j) {
      o0[j] *= alpha[j]; o1[j] *= alpha[j];
      o2[j] *= alpha[j]; o3[j] *= alpha[j];
      myp[(j + hi) * 32 +      (lane & 15)] = (bf16_t)p0[j];
      myp[(j + hi) * 32 + 16 + (lane & 15)] = (bf16_t)p1[j];
    }
    __syncthreads();

    // O += P @ V  (A-layout P from LDS, B = V^T rows = d)
    v16bf pf  = load_frag(myp, 32, 0, 0);
    v16bf vf0 = load_frag(vt, 2048, 0,  n);
    v16bf vf1 = load_frag(vt, 2048, 16, n);
    v16bf vf2 = load_frag(vt, 2048, 32, n);
    v16bf vf3 = load_frag(vt, 2048, 48, n);
    o0 = wmma_bf16(pf, vf0, o0);
    o1 = wmma_bf16(pf, vf1, o1);
    o2 = wmma_bf16(pf, vf2, o2);
    o3 = wmma_bf16(pf, vf3, o3);
    __syncthreads();
  }

#pragma unroll
  for (int j = 0; j < 8; ++j) {
    const float inv = 1.0f / lrun[j];
    const int seq = m0 + j + hi;
    const long row = ((long)b * 2048 + seq) * 768 + h * 64;
    AO[row +  0 + (lane & 15)] = (bf16_t)(o0[j] * inv);
    AO[row + 16 + (lane & 15)] = (bf16_t)(o1[j] * inv);
    AO[row + 32 + (lane & 15)] = (bf16_t)(o2[j] * inv);
    AO[row + 48 + (lane & 15)] = (bf16_t)(o3[j] * inv);
  }
}

// ---------------------------------------------------------------------------
// Output projection: out[m,n] = sum_k AO[m,k] * Wo[n,k] + bo[n]  (f32 out)
// Same 32x64 register blocking as the QKV GEMM.
// ---------------------------------------------------------------------------
__global__ __launch_bounds__(128) void mha_out_proj(
    const bf16_t* __restrict__ AO, const bf16_t* __restrict__ Wo,
    const float* __restrict__ bo, float* __restrict__ out) {
  const int wave = threadIdx.x >> 5;
  const int lane = threadIdx.x & 31;
  const int m0 = blockIdx.x * 32;
  const int n0 = (blockIdx.y * 4 + wave) * 64;

  v8f acc00 = {}, acc01 = {}, acc02 = {}, acc03 = {};
  v8f acc10 = {}, acc11 = {}, acc12 = {}, acc13 = {};

  for (int k = 0; k < 768; k += 32) {
    v16bf a0 = load_frag(AO, 768, m0,      k);
    v16bf a1 = load_frag(AO, 768, m0 + 16, k);
    v16bf b0 = load_frag(Wo, 768, n0,      k);
    v16bf b1 = load_frag(Wo, 768, n0 + 16, k);
    v16bf b2 = load_frag(Wo, 768, n0 + 32, k);
    v16bf b3 = load_frag(Wo, 768, n0 + 48, k);
    acc00 = wmma_bf16(a0, b0, acc00);
    acc01 = wmma_bf16(a0, b1, acc01);
    acc02 = wmma_bf16(a0, b2, acc02);
    acc03 = wmma_bf16(a0, b3, acc03);
    acc10 = wmma_bf16(a1, b0, acc10);
    acc11 = wmma_bf16(a1, b1, acc11);
    acc12 = wmma_bf16(a1, b2, acc12);
    acc13 = wmma_bf16(a1, b3, acc13);
  }

  const int hi = (lane & 16) ? 8 : 0;
  v8f* accs[2][4] = {{&acc00, &acc01, &acc02, &acc03},
                     {&acc10, &acc11, &acc12, &acc13}};
#pragma unroll
  for (int mi = 0; mi < 2; ++mi) {
#pragma unroll
    for (int ni = 0; ni < 4; ++ni) {
      const v8f a = *accs[mi][ni];
      const int n_g = n0 + ni * 16 + (lane & 15);
      const float bias = bo[n_g];
#pragma unroll
      for (int j = 0; j < 8; ++j) {
        const int m_g = m0 + mi * 16 + j + hi;
        out[(long)m_g * 768 + n_g] = a[j] + bias;
      }
    }
  }
}

// ---------------------------------------------------------------------------
// Host launcher
// ---------------------------------------------------------------------------
extern "C" void kernel_launch(void* const* d_in, const int* in_sizes, int n_in,
                              void* d_out, int out_size, void* d_ws, size_t ws_size,
                              hipStream_t stream) {
  const float* x    = (const float*)d_in[0];   // [2,2048,768]
  const int*   mask = (const int*)d_in[1];     // [2,2048]
  const float* Wq   = (const float*)d_in[2];   // [768,768]
  const float* Wk   = (const float*)d_in[3];
  const float* Wv   = (const float*)d_in[4];
  const float* Wo   = (const float*)d_in[5];
  const float* bo   = (const float*)d_in[6];   // [768]
  float* out = (float*)d_out;                  // [2,2048,768]

  const int M = 4096, C = 768;
  const size_t XB  = (size_t)M * C;            // 3,145,728 elems
  const size_t WB  = (size_t)C * C;            //   589,824 elems

  char* ws = (char*)d_ws;
  size_t off = 0;
  auto take = [&](size_t elems) { bf16_t* p = (bf16_t*)(ws + off);
                                  off += ((elems * 2 + 255) & ~(size_t)255); return p; };
  bf16_t* xb = take(XB);
  bf16_t* wq = take(WB);
  bf16_t* wk = take(WB);
  bf16_t* wv = take(WB);
  bf16_t* wo = take(WB);
  bf16_t* qb = take(XB);   // [bh][seq][64]
  bf16_t* kb = take(XB);   // [bh][seq][64]
  bf16_t* vt = take(XB);   // [bh][64][seq]
  bf16_t* ao = take(XB);   // [b][seq][768]

  // 1) converts
  mha_f32_to_bf16<<<(int)((XB + 255) / 256), 256, 0, stream>>>(x,  xb, (int)XB);
  mha_f32_to_bf16<<<(int)((WB + 255) / 256), 256, 0, stream>>>(Wq, wq, (int)WB);
  mha_f32_to_bf16<<<(int)((WB + 255) / 256), 256, 0, stream>>>(Wk, wk, (int)WB);
  mha_f32_to_bf16<<<(int)((WB + 255) / 256), 256, 0, stream>>>(Wv, wv, (int)WB);
  mha_f32_to_bf16<<<(int)((WB + 255) / 256), 256, 0, stream>>>(Wo, wo, (int)WB);

  // 2) QKV projections (M/32 = 128 blocks, N/64 = 12 blocks, 4 waves/block)
  dim3 gemm_grid(128, 3), gemm_blk(128);
  mha_qkv_gemm<<<gemm_grid, gemm_blk, 0, stream>>>(xb, wq, qb, 0);
  mha_qkv_gemm<<<gemm_grid, gemm_blk, 0, stream>>>(xb, wk, kb, 0);
  mha_qkv_gemm<<<gemm_grid, gemm_blk, 0, stream>>>(xb, wv, vt, 1);

  // 3) flash attention: 24 bh * 128 mtiles = 3072 waves = 768 blocks
  mha_flash_attn<<<768, 128, 0, stream>>>(qb, kb, vt, mask, ao);

  // 4) output projection + bias
  mha_out_proj<<<gemm_grid, gemm_blk, 0, stream>>>(ao, wo, bo, out);
}
